// QuantizedLinear_32727650795731
// MI455X (gfx1250) — matrix operand order repaired and checked by reading, not compiled, not run on hardware
//
#include <hip/hip_runtime.h>
#include <stdint.h>

typedef __attribute__((ext_vector_type(16))) __bf16         v16bf;
typedef __attribute__((ext_vector_type(8)))  float          v8f;
typedef __attribute__((ext_vector_type(4)))  float          v4f;
typedef __attribute__((ext_vector_type(16))) unsigned short v16u;
typedef __attribute__((ext_vector_type(8)))  unsigned short v8u;
typedef __attribute__((ext_vector_type(4)))  unsigned int   v4u;

#define IN_F  4096
#define OUT_F 4096
#define BM 128
#define BN 256
#define BK 64

#if defined(__AMDGCN__) && \
    __has_builtin(__builtin_amdgcn_global_load_async_to_lds_b128) && \
    __has_builtin(__builtin_amdgcn_s_wait_asynccnt)
#define HAVE_ASYNC_LDS 1
// exact pointee type the builtin expects: GCC-style int4 vector
typedef int i4_t __attribute__((vector_size(16)));
typedef __attribute__((address_space(1))) i4_t gi4;   // global (AS1)
typedef __attribute__((address_space(3))) i4_t li4;   // LDS (AS3)
#else
#define HAVE_ASYNC_LDS 0
#endif

// f32 -> bf16 bits, round-to-nearest-even
__device__ __forceinline__ unsigned short f2bf(float f) {
    unsigned int u = __builtin_bit_cast(unsigned int, f);
    unsigned int r = u + 0x7FFFu + ((u >> 16) & 1u);
    return (unsigned short)(r >> 16);
}
__device__ __forceinline__ unsigned int pack2(float a, float b) {
    return (unsigned int)f2bf(a) | ((unsigned int)f2bf(b) << 16);
}

// 16-byte global -> LDS copy (async DMA when available)
__device__ __forceinline__ void copy16(const unsigned short* g, unsigned short* l) {
#if HAVE_ASYNC_LDS
    __builtin_amdgcn_global_load_async_to_lds_b128(
        (gi4*)(i4_t*)(void*)(unsigned short*)g,   // generic -> AS1, int4 pointee
        (li4*)(i4_t*)(void*)l,                    // generic -> AS3, int4 pointee
        0, 0);
#else
    *(v4u*)l = *(const v4u*)g;
#endif
}
__device__ __forceinline__ void copy_fence() {
#if HAVE_ASYNC_LDS
    __builtin_amdgcn_s_wait_asynccnt(0);
#endif
}

// ---------- prepass 1: x f32 -> bf16 ----------
__global__ __launch_bounds__(256)
void cvt_x_bf16(const v4f* __restrict__ x, v4u* __restrict__ xb) {
    size_t i = (size_t)blockIdx.x * 256 + threadIdx.x;   // one 16B bf16 chunk (8 vals)
    v4f a = __builtin_nontemporal_load(&x[2 * i]);
    v4f b = __builtin_nontemporal_load(&x[2 * i + 1]);
    v4u r = (v4u){pack2(a.x, a.y), pack2(a.z, a.w), pack2(b.x, b.y), pack2(b.z, b.w)};
    __builtin_nontemporal_store(r, &xb[i]);
}

// ---------- prepass 2: int4 -> bf16 dequant, row-major [OUT_F][IN_F] ----------
__global__ __launch_bounds__(256)
void dequant_w_bf16(const unsigned char* __restrict__ pw, unsigned short* __restrict__ wb,
                    const float* __restrict__ scale_p, const float* __restrict__ zp_p) {
    const float scale = *scale_p;
    const float nzp   = -scale * (*zp_p);            // W = q*scale + nzp
    const int t    = blockIdx.x * 256 + threadIdx.x; // 16 bytes each
    const int brow = t >> 8;                         // byte row (0..2047)
    const int kc   = (t & 255) * 16;                 // k offset
    const v4u wv = *(const v4u*)(pw + (size_t)brow * IN_F + kc);

    unsigned int ep[8], op[8];
#pragma unroll
    for (int d = 0; d < 4; ++d) {
        unsigned int wd = wv[d];
        float l0 = fmaf((float)( wd        & 0xFu), scale, nzp);
        float h0 = fmaf((float)((wd >> 4)  & 0xFu), scale, nzp);
        float l1 = fmaf((float)((wd >> 8)  & 0xFu), scale, nzp);
        float h1 = fmaf((float)((wd >> 12) & 0xFu), scale, nzp);
        float l2 = fmaf((float)((wd >> 16) & 0xFu), scale, nzp);
        float h2 = fmaf((float)((wd >> 20) & 0xFu), scale, nzp);
        float l3 = fmaf((float)((wd >> 24) & 0xFu), scale, nzp);
        float h3 = fmaf((float)((wd >> 28) & 0xFu), scale, nzp);
        ep[2 * d]     = pack2(l0, l1);
        ep[2 * d + 1] = pack2(l2, l3);
        op[2 * d]     = pack2(h0, h1);
        op[2 * d + 1] = pack2(h2, h3);
    }
    v4u* re = (v4u*)(wb + (size_t)(2 * brow)     * IN_F + kc);
    v4u* ro = (v4u*)(wb + (size_t)(2 * brow + 1) * IN_F + kc);
    __builtin_nontemporal_store((v4u){ep[0], ep[1], ep[2], ep[3]}, re);
    __builtin_nontemporal_store((v4u){ep[4], ep[5], ep[6], ep[7]}, re + 1);
    __builtin_nontemporal_store((v4u){op[0], op[1], op[2], op[3]}, ro);
    __builtin_nontemporal_store((v4u){op[4], op[5], op[6], op[7]}, ro + 1);
}

// ---------- main: bf16 GEMM, 128x256 block, 64x64 wave tiles, async double buffer ----------
__global__ __launch_bounds__(256)
void gemm_bf16_wmma(const unsigned short* __restrict__ xb,
                    const unsigned short* __restrict__ wb,
                    const float* __restrict__ bias,
                    float* __restrict__ out) {
    __shared__ unsigned short Als[2][BM][BK];   // 32 KB
    __shared__ unsigned short Bls[2][BN][BK];   // 64 KB

    const int tid  = threadIdx.x;
    const int lane = tid & 31;
    const int wave = tid >> 5;
    const int wm   = wave & 1;                  // 2 slabs of 64 rows
    const int wn   = wave >> 1;                 // 4 slabs of 64 cols
    const int m0   = blockIdx.y * BM;
    const int n0   = blockIdx.x * BN;

    v8f acc[4][4];
#pragma unroll
    for (int mi = 0; mi < 4; ++mi)
#pragma unroll
        for (int ni = 0; ni < 4; ++ni)
            acc[mi][ni] = (v8f){0.f, 0.f, 0.f, 0.f, 0.f, 0.f, 0.f, 0.f};

    // tile fill: A = 1024 16B chunks (4/thread), B = 2048 (8/thread)
    const int crow = tid >> 3;        // base row; +32 per chunk pass
    const int ccol = (tid & 7) * 8;   // halves within row

    // prologue: fill buffer 0 with k0 = 0
#pragma unroll
    for (int q = 0; q < 4; ++q)
        copy16(xb + (size_t)(m0 + crow + 32 * q) * IN_F + ccol, &Als[0][crow + 32 * q][ccol]);
#pragma unroll
    for (int q = 0; q < 8; ++q)
        copy16(wb + (size_t)(n0 + crow + 32 * q) * IN_F + ccol, &Bls[0][crow + 32 * q][ccol]);

    for (int k0 = 0; k0 < IN_F; k0 += BK) {
        const int cur = (k0 >> 6) & 1;
        copy_fence();          // this wave's async fills for `cur` done
        __syncthreads();       // all waves' fills visible; prev reads finished

        if (k0 + BK < IN_F) {  // kick off next tile into the other buffer
            const int nxt = cur ^ 1;
            const int kn  = k0 + BK;
#pragma unroll
            for (int q = 0; q < 4; ++q)
                copy16(xb + (size_t)(m0 + crow + 32 * q) * IN_F + kn + ccol,
                       &Als[nxt][crow + 32 * q][ccol]);
#pragma unroll
            for (int q = 0; q < 8; ++q)
                copy16(wb + (size_t)(n0 + crow + 32 * q) * IN_F + kn + ccol,
                       &Bls[nxt][crow + 32 * q][ccol]);
        }

        // compute on `cur`: 2 K-steps of 32, 16 WMMA each
#pragma unroll
        for (int ks = 0; ks < BK; ks += 32) {
            const int mlane = lane & 15;
            const int klo   = (lane < 16) ? 0 : 16;   // A: K 0..15 vs 16..31
            const int kb    = ks + ((lane < 16) ? 0 : 16); // B: K 0..15 vs 16..31

            v16bf bfr[4];
#pragma unroll
            for (int ni = 0; ni < 4; ++ni) {
                const unsigned short* bp = &Bls[cur][wn * 64 + ni * 16 + mlane][kb];
                bfr[ni] = __builtin_bit_cast(v16bf, *(const v16u*)bp);
            }

#pragma unroll
            for (int mi = 0; mi < 4; ++mi) {
                const unsigned short* ap = &Als[cur][wm * 64 + mi * 16 + mlane][ks + klo];
                v8u lo = *(const v8u*)ap;
                v8u hi = *(const v8u*)(ap + 8);
                v16u comb = __builtin_shufflevector(lo, hi,
                    0, 1, 2, 3, 4, 5, 6, 7, 8, 9, 10, 11, 12, 13, 14, 15);
                v16bf afr = __builtin_bit_cast(v16bf, comb);
#pragma unroll
                for (int ni = 0; ni < 4; ++ni)
                    acc[mi][ni] = __builtin_amdgcn_wmma_f32_16x16x32_bf16(
                        false, afr, false, bfr[ni],
                        (short)0, acc[mi][ni], false, false);
            }
        }
    }

    // epilogue: bias add + NT store (C layout: VGPR r -> M=r / M=r+8 at lane 16)
    const int nc      = lane & 15;
    const int colbase = n0 + wn * 64;
    const int rowbase = m0 + wm * 64 + ((lane >= 16) ? 8 : 0);
#pragma unroll
    for (int ni = 0; ni < 4; ++ni) {
        const int col = colbase + ni * 16 + nc;
        const float bv = bias[col];
#pragma unroll
        for (int mi = 0; mi < 4; ++mi) {
#pragma unroll
            for (int r = 0; r < 8; ++r) {
                const int row = rowbase + mi * 16 + r;
                __builtin_nontemporal_store(acc[mi][ni][r] + bv,
                                            out + (size_t)row * OUT_F + col);
            }
        }
    }
}

extern "C" void kernel_launch(void* const* d_in, const int* in_sizes, int n_in,
                              void* d_out, int out_size, void* d_ws, size_t ws_size,
                              hipStream_t stream) {
    const float*         x     = (const float*)d_in[0];
    const unsigned char* pw    = (const unsigned char*)d_in[1];
    const float*         scale = (const float*)d_in[2];
    const float*         zp    = (const float*)d_in[3];
    const float*         bias  = (const float*)d_in[4];
    float*               out   = (float*)d_out;

    const int M = in_sizes[0] / IN_F;                       // 8192
    unsigned short* xb = (unsigned short*)d_ws;             // M*IN_F bf16
    unsigned short* wb = xb + (size_t)M * IN_F;             // OUT_F*IN_F bf16

    // prepass 1: x -> bf16   (M*IN_F/8 chunks)
    cvt_x_bf16<<<(unsigned)((size_t)M * IN_F / 8 / 256), 256, 0, stream>>>(
        (const v4f*)x, (v4u*)xb);
    // prepass 2: dequant weights -> bf16  (OUT_F/2 * IN_F / 16 chunks)
    dequant_w_bf16<<<(unsigned)((size_t)(OUT_F / 2) * IN_F / 16 / 256), 256, 0, stream>>>(
        pw, wb, scale, zp);
    // main GEMM
    dim3 grid(OUT_F / BN, M / BM);                          // (16, 64)
    gemm_bf16_wmma<<<grid, 256, 0, stream>>>(xb, wb, bias, out);
}
